// CrossAttentionTransformer_2173253451852
// MI455X (gfx1250) — matrix operand order repaired and checked by reading, not compiled, not run on hardware
//
#include <hip/hip_runtime.h>
#include <hip/hip_bf16.h>
#include <math.h>

// ---------------------------------------------------------------------------
// CrossAttentionTransformer layer for MI455X (gfx1250, wave32, WMMA bf16)
// Dense projections/MLPs -> v_wmma_f32_16x16x32_bf16 (fp32 accum), weight
// panels staged in LDS via GLOBAL_LOAD_ASYNC_TO_LDS (ASYNCcnt) when available.
// Sparse edge attention -> 3-pass segment softmax with f32 global atomics.
// ---------------------------------------------------------------------------

#define NN   60882      // N_NODE
#define NTRI 115443     // N_TRI
#define DD   256
#define DFF  1024
#define EN   365292     // 6*N_NODE
#define ET   346329     // 3*N_TRI
#define EC   346329     // 3*N_TRI

typedef __attribute__((ext_vector_type(16))) __bf16 v16bf;
typedef __attribute__((ext_vector_type(8)))  __bf16 v8bf;
typedef __attribute__((ext_vector_type(8)))  float  v8f;
typedef __attribute__((__vector_size__(4 * sizeof(int)))) int vi4;  // async-lds builtin arg type

#if defined(__has_builtin)
#if __has_builtin(__builtin_amdgcn_global_load_async_to_lds_b128) && \
    __has_builtin(__builtin_amdgcn_s_wait_asynccnt)
#define USE_ASYNC_LDS 1
#endif
#endif

__device__ __forceinline__ float bf2f(unsigned int u) {
  return __uint_as_float(u << 16);
}
__device__ __forceinline__ unsigned short f2bf(float f) {
  unsigned u = __float_as_uint(f);
  u += 0x7fffu + ((u >> 16) & 1u);      // round-to-nearest-even
  return (unsigned short)(u >> 16);
}

// ------------------------------ utility kernels ----------------------------

__global__ void fill_f32(float* __restrict__ p, float v, long n) {
  long i = (long)blockIdx.x * blockDim.x + threadIdx.x;
  long s = (long)gridDim.x * blockDim.x;
  for (; i < n; i += s) p[i] = v;
}

__global__ void add_f32(const float* __restrict__ a, const float* __restrict__ b,
                        float* __restrict__ o, long n) {
  long i = (long)blockIdx.x * blockDim.x + threadIdx.x;
  long s = (long)gridDim.x * blockDim.x;
  for (; i < n; i += s) o[i] = a[i] + b[i];
}

__global__ void f32_to_bf16_k(const float* __restrict__ a,
                              unsigned short* __restrict__ o, long n) {
  long i = (long)blockIdx.x * blockDim.x + threadIdx.x;
  long s = (long)gridDim.x * blockDim.x;
  for (; i < n; i += s) o[i] = f2bf(a[i]);
}

// ------------------------------ layernorm ----------------------------------
// One wave32 per row of 256; 8 channels per lane. gamma/beta staged in LDS.

__global__ void ln_kernel(const float* __restrict__ x, const float* __restrict__ g,
                          const float* __restrict__ b,
                          unsigned short* __restrict__ outb,
                          float* __restrict__ outf, int N) {
  __shared__ float sg[DD];
  __shared__ float sb[DD];
  for (int i = threadIdx.x; i < DD; i += blockDim.x) { sg[i] = g[i]; sb[i] = b[i]; }
  __syncthreads();

  int row = blockIdx.x * (blockDim.x >> 5) + (threadIdx.x >> 5);
  if (row >= N) return;
  int lane = threadIdx.x & 31;

  const float4* xp = (const float4*)(x + (size_t)row * DD + lane * 8);
  float4 a0 = xp[0], a1 = xp[1];
  float v[8] = {a0.x, a0.y, a0.z, a0.w, a1.x, a1.y, a1.z, a1.w};

  float s = 0.f, s2 = 0.f;
#pragma unroll
  for (int i = 0; i < 8; ++i) { s += v[i]; s2 += v[i] * v[i]; }
  for (int off = 16; off; off >>= 1) {
    s  += __shfl_xor(s,  off, 32);
    s2 += __shfl_xor(s2, off, 32);
  }
  float mu   = s * (1.0f / DD);
  float var  = s2 * (1.0f / DD) - mu * mu;
  float rstd = rsqrtf(var + 1e-5f);

  alignas(16) unsigned short ob[8];
  alignas(16) float of[8];
#pragma unroll
  for (int i = 0; i < 8; ++i) {
    int c = lane * 8 + i;
    float y = (v[i] - mu) * rstd * sg[c] + sb[c];
    of[i] = y;
    ob[i] = f2bf(y);
  }
  *(uint4*)(outb + (size_t)row * DD + lane * 8) = *(uint4*)ob;
  if (outf) {
    float4* op = (float4*)(outf + (size_t)row * DD + lane * 8);
    op[0] = *(float4*)of;
    op[1] = *(float4*)(of + 4);
  }
}

// ------------------------------ WMMA GEMM ----------------------------------
// y[M,Nout] = x[M,K](bf16) @ w[Nout,K]^T (bf16) + bias, fused epilogue.
// MODE 0: f32 out  (+ optional res1/res2 residual adds)
// MODE 1: bf16 out
// MODE 2: bf16 out with exact GELU
//
// Block = 8 waves = 128 rows x 64 cols. The 64-row weight panel (contiguous
// in memory since w is [out,in] row-major) is staged into LDS, chunked at
// KC<=512 so the panel is at most 64KB; async copy to LDS when the toolchain
// exposes GLOBAL_LOAD_ASYNC_TO_LDS, else load+ds_store. Each wave then runs
// 1 A-fragment (global, streaming) x 4 B-fragments (ds_load) per K-step.
//
// Fragment layouts per CDNA5 ISA 7.12.2:
//   A (16x32 bf16): lane m=l&15; lanes<16 carry K {0..7,16..23}, lanes>=16
//                   carry K {8..15,24..31}  -> two 16B contiguous loads.
//   B (32x16 bf16): lane n=l&15; lanes<16 carry K {0..7,16..23}, lanes>=16
//                   carry K {8..15,24..31}  -> two 16B contiguous loads.
//   D (16x16 f32):  lane n=l&15; vgpr j -> m = j + (l>=16 ? 8 : 0).

template <int MODE>
__global__ void gemm_wmma(const unsigned short* __restrict__ A,
                          const unsigned short* __restrict__ W,
                          const float* __restrict__ bias,
                          const float* __restrict__ res1,
                          const float* __restrict__ res2,
                          float* __restrict__ outf,
                          unsigned short* __restrict__ outb,
                          int M, int K, int Nout) {
  extern __shared__ unsigned short sW[];   // [64][KC] panel of W (bf16)

  const int lane = threadIdx.x & 31;
  const int wave = threadIdx.x >> 5;
  const int m0 = (blockIdx.x * 8 + wave) * 16;
  const int n0 = blockIdx.y * 64;

  int mA = m0 + (lane & 15);
  if (mA > M - 1) mA = M - 1;               // clamp loads; stores are guarded
  const int kbA = (lane < 16) ? 0 : 8;
  const unsigned short* Ap = A + (size_t)mA * K + kbA;

  const int nBl = lane & 15;
  const int kbB = (lane < 16) ? 0 : 8;

  const int KC   = (K > 512) ? 512 : K;     // K-chunk staged in LDS
  const int kcsh = (KC == 512) ? 9 : 8;     // log2(KC); KC in {256,512}
  const int total8 = (64 * KC) >> 3;        // 16B chunks in panel

  v8f acc0 = {}, acc1 = {}, acc2 = {}, acc3 = {};

  for (int kc = 0; kc < K; kc += KC) {
    // ---- stage W[n0..n0+63][kc..kc+KC) into LDS ----
    for (int j = threadIdx.x; j < total8; j += 256) {
      int jj = j << 3;
      int r = jj >> kcsh;
      int c = jj & (KC - 1);
      const unsigned short* srcp = W + (size_t)(n0 + r) * K + kc + c;
      unsigned short* dstp = sW + jj;
#ifdef USE_ASYNC_LDS
      __builtin_amdgcn_global_load_async_to_lds_b128((vi4*)srcp, (vi4*)dstp, 0, 0);
#else
      *(uint4*)dstp = *(const uint4*)srcp;
#endif
    }
#ifdef USE_ASYNC_LDS
    __builtin_amdgcn_s_wait_asynccnt(0);
#endif
    __syncthreads();

    for (int k0 = 0; k0 < KC; k0 += 32) {
      v16bf af;
      {
        const unsigned short* ap = Ap + kc + k0;
        v8bf lo = *(const v8bf*)ap;
        v8bf hi = *(const v8bf*)(ap + 16);
#pragma unroll
        for (int i = 0; i < 8; ++i) { af[i] = lo[i]; af[8 + i] = hi[i]; }
        if (k0 + 32 < KC) __builtin_prefetch(ap + 32, 0, 0);
      }
#pragma unroll
      for (int t = 0; t < 4; ++t) {
        const unsigned short* Wp = sW + (size_t)(nBl + t * 16) * KC + k0 + kbB;
        v16bf bfm;
        v8bf lo = *(const v8bf*)Wp;
        v8bf hi = *(const v8bf*)(Wp + 16);
#pragma unroll
        for (int i = 0; i < 8; ++i) { bfm[i] = lo[i]; bfm[8 + i] = hi[i]; }
        v8f& acc = (t == 0 ? acc0 : t == 1 ? acc1 : t == 2 ? acc2 : acc3);
        acc = __builtin_amdgcn_wmma_f32_16x16x32_bf16(
            false, af, false, bfm, (short)0, acc, false, false);
      }
    }
    __syncthreads();
  }

  if (m0 >= M) return;

#pragma unroll
  for (int t = 0; t < 4; ++t) {
    v8f acc = (t == 0 ? acc0 : t == 1 ? acc1 : t == 2 ? acc2 : acc3);
    int n = n0 + t * 16 + (lane & 15);
    float bn = bias[n];
#pragma unroll
    for (int j = 0; j < 8; ++j) {
      int m = m0 + j + ((lane >= 16) ? 8 : 0);
      if (m < M) {
        size_t idx = (size_t)m * Nout + n;
        float val = acc[j] + bn;
        if (MODE == 0) {
          if (res1) val += res1[idx];
          if (res2) val += res2[idx];
          outf[idx] = val;
        } else {
          if (MODE == 2) val = 0.5f * val * (1.0f + erff(val * 0.70710678118f));
          outb[idx] = f2bf(val);
        }
      }
    }
  }
}

// ------------------------------ edge attention -----------------------------

__device__ __forceinline__ void atomicMaxF32(float* addr, float val) {
  if (val >= 0.0f) atomicMax((int*)addr, __float_as_int(val));
  else             atomicMin((unsigned int*)addr, __float_as_uint(val));
}

// pass 1: logits[e] = scale * dot(q[dst], k[src]); segment atomic max
__global__ void edge_logits_max(const unsigned short* __restrict__ q,
                                const unsigned short* __restrict__ k,
                                const int* __restrict__ src,
                                const int* __restrict__ dst,
                                int E, float scale,
                                float* __restrict__ logits,
                                float* __restrict__ smax) {
  int e = blockIdx.x * (blockDim.x >> 5) + (threadIdx.x >> 5);
  if (e >= E) return;
  int lane = threadIdx.x & 31;
  int si = src[e], di = dst[e];
  uint4 qv = *(const uint4*)(q + (size_t)di * DD + lane * 8);
  uint4 kv = *(const uint4*)(k + (size_t)si * DD + lane * 8);
  float s =
      bf2f(qv.x & 0xffffu) * bf2f(kv.x & 0xffffu) + bf2f(qv.x >> 16) * bf2f(kv.x >> 16) +
      bf2f(qv.y & 0xffffu) * bf2f(kv.y & 0xffffu) + bf2f(qv.y >> 16) * bf2f(kv.y >> 16) +
      bf2f(qv.z & 0xffffu) * bf2f(kv.z & 0xffffu) + bf2f(qv.z >> 16) * bf2f(kv.z >> 16) +
      bf2f(qv.w & 0xffffu) * bf2f(kv.w & 0xffffu) + bf2f(qv.w >> 16) * bf2f(kv.w >> 16);
  for (int off = 16; off; off >>= 1) s += __shfl_xor(s, off, 32);
  if (lane == 0) {
    float l = s * scale;
    logits[e] = l;
    atomicMaxF32(&smax[di], l);
  }
}

// pass 2: ex = exp(logit - max[dst]); segment atomic sum (ex overwrites logits)
__global__ void edge_exp_sum(float* __restrict__ logits,
                             const int* __restrict__ dst,
                             const float* __restrict__ smax,
                             float* __restrict__ ssum, int E) {
  int e = blockIdx.x * blockDim.x + threadIdx.x;
  if (e >= E) return;
  int di = dst[e];
  float ex = expf(logits[e] - smax[di]);
  logits[e] = ex;
  atomicAdd(&ssum[di], ex);
}

// pass 3: aggr[dst] += (ex / (sum[dst]+eps)) * v[src]
__global__ void edge_aggr(const float* __restrict__ ex,
                          const float* __restrict__ ssum,
                          const unsigned short* __restrict__ v,
                          const int* __restrict__ src,
                          const int* __restrict__ dst,
                          int E, float* __restrict__ aggr) {
  int e = blockIdx.x * (blockDim.x >> 5) + (threadIdx.x >> 5);
  if (e >= E) return;
  int lane = threadIdx.x & 31;
  int si = src[e], di = dst[e];
  float alpha = ex[e] / (ssum[di] + 1e-16f);
  uint4 vv = *(const uint4*)(v + (size_t)si * DD + lane * 8);
  float* ap = aggr + (size_t)di * DD + lane * 8;
  atomicAdd(ap + 0, alpha * bf2f(vv.x & 0xffffu));
  atomicAdd(ap + 1, alpha * bf2f(vv.x >> 16));
  atomicAdd(ap + 2, alpha * bf2f(vv.y & 0xffffu));
  atomicAdd(ap + 3, alpha * bf2f(vv.y >> 16));
  atomicAdd(ap + 4, alpha * bf2f(vv.z & 0xffffu));
  atomicAdd(ap + 5, alpha * bf2f(vv.z >> 16));
  atomicAdd(ap + 6, alpha * bf2f(vv.w & 0xffffu));
  atomicAdd(ap + 7, alpha * bf2f(vv.w >> 16));
}

// ------------------------------ host orchestration -------------------------

extern "C" void kernel_launch(void* const* d_in, const int* in_sizes, int n_in,
                              void* d_out, int out_size, void* d_ws, size_t ws_size,
                              hipStream_t stream) {
  (void)in_sizes; (void)n_in; (void)out_size; (void)ws_size;

  const float* node0 = (const float*)d_in[0];
  const float* tri0  = (const float*)d_in[1];
  const int* ne = (const int*)d_in[2];   // [2, EN]  (src, dst)
  const int* te = (const int*)d_in[3];   // [2, ET]
  const int* tn = (const int*)d_in[4];   // [2, EC]  src: tri,  dst: node
  const int* nt = (const int*)d_in[5];   // [2, EC]  src: node, dst: tri

  // params flattened in jax pytree (sorted dict key) order:
  //  ln_node_cross{b,g} ln_node_mlp{b,g} ln_node_self{b,g}
  //  ln_tri_cross{b,g}  ln_tri_mlp{b,g}  ln_tri_self{b,g}
  //  mlp_node{b1,b2,w1,w2} mlp_tri{b1,b2,w1,w2}
  //  node2tri{kb,kw,ob,ow,qb,qw,vb,vw}
  //  node_sa{kb,kw,qb,qw,vb,vw}
  //  tri2node{kb,kw,ob,ow,qb,qw,vb,vw}
  //  tri_sa{kb,kw,qb,qw,vb,vw}
  auto P = [&](int i) { return (const float*)d_in[6 + i]; };
  const float *ln_nc_b = P(0),  *ln_nc_g = P(1);
  const float *ln_nm_b = P(2),  *ln_nm_g = P(3);
  const float *ln_ns_b = P(4),  *ln_ns_g = P(5);
  const float *ln_tc_b = P(6),  *ln_tc_g = P(7);
  const float *ln_tm_b = P(8),  *ln_tm_g = P(9);
  const float *ln_ts_b = P(10), *ln_ts_g = P(11);
  const float *mn_b1 = P(12), *mn_b2 = P(13), *mn_w1 = P(14), *mn_w2 = P(15);
  const float *mt_b1 = P(16), *mt_b2 = P(17), *mt_w1 = P(18), *mt_w2 = P(19);
  const float *n2t_kb = P(20), *n2t_kw = P(21), *n2t_ob = P(22), *n2t_ow = P(23),
              *n2t_qb = P(24), *n2t_qw = P(25), *n2t_vb = P(26), *n2t_vw = P(27);
  const float *nsa_kb = P(28), *nsa_kw = P(29), *nsa_qb = P(30), *nsa_qw = P(31),
              *nsa_vb = P(32), *nsa_vw = P(33);
  const float *t2n_kb = P(34), *t2n_kw = P(35), *t2n_ob = P(36), *t2n_ow = P(37),
              *t2n_qb = P(38), *t2n_qw = P(39), *t2n_vb = P(40), *t2n_vw = P(41);
  const float *tsa_kb = P(42), *tsa_kw = P(43), *tsa_qb = P(44), *tsa_qw = P(45),
              *tsa_vb = P(46), *tsa_vw = P(47);

  float* out_node = (float*)d_out;
  float* out_tri  = out_node + (size_t)NN * DD;

  // workspace bump allocator
  char* ws = (char*)d_ws;
  size_t cur = 0;
  auto alloc = [&](size_t bytes) -> void* {
    cur = (cur + 255) & ~(size_t)255;
    void* p = ws + cur;
    cur += bytes;
    return p;
  };
  unsigned short* LNSN = (unsigned short*)alloc((size_t)NN   * DD * 2);  // ln_node_self bf16
  unsigned short* LNST = (unsigned short*)alloc((size_t)NTRI * DD * 2);  // ln_tri_self bf16
  unsigned short* Qb   = (unsigned short*)alloc((size_t)NTRI * DD * 2);
  unsigned short* Kb   = (unsigned short*)alloc((size_t)NTRI * DD * 2);
  unsigned short* Vb   = (unsigned short*)alloc((size_t)NTRI * DD * 2);
  unsigned short* XLNb = (unsigned short*)alloc((size_t)NTRI * DD * 2);  // ln cross/mlp bf16
  float* XLNf = (float*)alloc((size_t)NTRI * DD * 4);                    // ln cross f32 (residual)
  float* AGG  = (float*)alloc((size_t)NTRI * DD * 4);                    // attention aggregate
  unsigned short* Hb = (unsigned short*)alloc((size_t)NTRI * DFF * 2);   // mlp hidden bf16
  float* LOG  = (float*)alloc((size_t)EN * 4);
  float* SMAX = (float*)alloc((size_t)NTRI * 4);
  float* SSUM = (float*)alloc((size_t)NTRI * 4);

  auto cvtw = [&](const float* w, size_t n) -> unsigned short* {
    unsigned short* p = (unsigned short*)alloc(n * 2);
    f32_to_bf16_k<<<1024, 256, 0, stream>>>(w, p, (long)n);
    return p;
  };
  const size_t WDD = (size_t)DD * DD;
  unsigned short *w_nsa_q = cvtw(nsa_qw, WDD), *w_nsa_k = cvtw(nsa_kw, WDD),
                 *w_nsa_v = cvtw(nsa_vw, WDD);
  unsigned short *w_tsa_q = cvtw(tsa_qw, WDD), *w_tsa_k = cvtw(tsa_kw, WDD),
                 *w_tsa_v = cvtw(tsa_vw, WDD);
  unsigned short *w_t2n_q = cvtw(t2n_qw, WDD), *w_t2n_k = cvtw(t2n_kw, WDD),
                 *w_t2n_v = cvtw(t2n_vw, WDD), *w_t2n_o = cvtw(t2n_ow, WDD);
  unsigned short *w_n2t_q = cvtw(n2t_qw, WDD), *w_n2t_k = cvtw(n2t_kw, WDD),
                 *w_n2t_v = cvtw(n2t_vw, WDD), *w_n2t_o = cvtw(n2t_ow, WDD);
  unsigned short *w_mn_1 = cvtw(mn_w1, (size_t)DFF * DD),
                 *w_mn_2 = cvtw(mn_w2, (size_t)DD * DFF);
  unsigned short *w_mt_1 = cvtw(mt_w1, (size_t)DFF * DD),
                 *w_mt_2 = cvtw(mt_w2, (size_t)DD * DFF);

  auto ln = [&](const float* x, const float* g, const float* b,
                unsigned short* ob, float* of, int N) {
    ln_kernel<<<dim3((N + 7) / 8), dim3(256), 0, stream>>>(x, g, b, ob, of, N);
  };
  auto gemm = [&](const unsigned short* A, const unsigned short* W, const float* bias,
                  int M, int K, int Nout, int mode, const float* r1, const float* r2,
                  float* of, unsigned short* ob) {
    dim3 grd((M + 127) / 128, Nout / 64);
    dim3 blk(256);
    size_t shmem = (size_t)64 * ((K > 512) ? 512 : K) * 2;  // W panel in LDS
    if (mode == 0)
      gemm_wmma<0><<<grd, blk, shmem, stream>>>(A, W, bias, r1, r2, of, ob, M, K, Nout);
    else if (mode == 1)
      gemm_wmma<1><<<grd, blk, shmem, stream>>>(A, W, bias, r1, r2, of, ob, M, K, Nout);
    else
      gemm_wmma<2><<<grd, blk, shmem, stream>>>(A, W, bias, r1, r2, of, ob, M, K, Nout);
  };
  auto attn = [&](const unsigned short* q, const unsigned short* k, const unsigned short* v,
                  const int* src, const int* dst, int E, int Ndst, float scale) {
    fill_f32<<<512,  256, 0, stream>>>(SMAX, -INFINITY, (long)Ndst);
    fill_f32<<<512,  256, 0, stream>>>(SSUM, 0.0f, (long)Ndst);
    fill_f32<<<2048, 256, 0, stream>>>(AGG, 0.0f, (long)Ndst * DD);
    edge_logits_max<<<(E + 7) / 8, 256, 0, stream>>>(q, k, src, dst, E, scale, LOG, SMAX);
    edge_exp_sum<<<(E + 255) / 256, 256, 0, stream>>>(LOG, dst, SMAX, SSUM, E);
    edge_aggr<<<(E + 7) / 8, 256, 0, stream>>>(LOG, SSUM, v, src, dst, E, AGG);
  };

  const float SC = 0.0625f;  // 256^-0.5
  const long NNE = (long)NN * DD, NTE = (long)NTRI * DD;

  // LN of originals (both kept: LNSN feeds node2tri k/v, LNST feeds tri2node k/v)
  ln(node0, ln_ns_g, ln_ns_b, LNSN, nullptr, NN);
  ln(tri0,  ln_ts_g, ln_ts_b, LNST, nullptr, NTRI);

  // ---- node self-attention (scale d^-0.5) ----
  gemm(LNSN, w_nsa_q, nsa_qb, NN, DD, DD, 1, nullptr, nullptr, nullptr, Qb);
  gemm(LNSN, w_nsa_k, nsa_kb, NN, DD, DD, 1, nullptr, nullptr, nullptr, Kb);
  gemm(LNSN, w_nsa_v, nsa_vb, NN, DD, DD, 1, nullptr, nullptr, nullptr, Vb);
  attn(Qb, Kb, Vb, ne, ne + EN, EN, NN, SC);
  add_f32<<<2048, 256, 0, stream>>>(node0, AGG, out_node, NNE);

  // ---- tri -> node cross attention (residual adds BOTH node and ln(node)) ----
  ln(out_node, ln_nc_g, ln_nc_b, XLNb, XLNf, NN);
  gemm(XLNb, w_t2n_q, t2n_qb, NN,   DD, DD, 1, nullptr, nullptr, nullptr, Qb);
  gemm(LNST, w_t2n_k, t2n_kb, NTRI, DD, DD, 1, nullptr, nullptr, nullptr, Kb);
  gemm(LNST, w_t2n_v, t2n_vb, NTRI, DD, DD, 1, nullptr, nullptr, nullptr, Vb);
  attn(Qb, Kb, Vb, tn, tn + EC, EC, NN, SC);
  f32_to_bf16_k<<<1024, 256, 0, stream>>>(AGG, Qb, NNE);
  gemm(Qb, w_t2n_o, t2n_ob, NN, DD, DD, 0, out_node, XLNf, out_node, nullptr);

  // ---- node MLP ----
  ln(out_node, ln_nm_g, ln_nm_b, XLNb, nullptr, NN);
  gemm(XLNb, w_mn_1, mn_b1, NN, DD,  DFF, 2, nullptr, nullptr, nullptr, Hb);
  gemm(Hb,   w_mn_2, mn_b2, NN, DFF, DD,  0, out_node, nullptr, out_node, nullptr);

  // ---- triangle self-attention (scale d^+0.5, faithful to reference) ----
  gemm(LNST, w_tsa_q, tsa_qb, NTRI, DD, DD, 1, nullptr, nullptr, nullptr, Qb);
  gemm(LNST, w_tsa_k, tsa_kb, NTRI, DD, DD, 1, nullptr, nullptr, nullptr, Kb);
  gemm(LNST, w_tsa_v, tsa_vb, NTRI, DD, DD, 1, nullptr, nullptr, nullptr, Vb);
  attn(Qb, Kb, Vb, te, te + ET, ET, NTRI, 16.0f);
  add_f32<<<2048, 256, 0, stream>>>(tri0, AGG, out_tri, NTE);

  // ---- node -> tri cross attention (k/v from ln of ORIGINAL node) ----
  ln(out_tri, ln_tc_g, ln_tc_b, XLNb, XLNf, NTRI);
  gemm(XLNb, w_n2t_q, n2t_qb, NTRI, DD, DD, 1, nullptr, nullptr, nullptr, Qb);
  gemm(LNSN, w_n2t_k, n2t_kb, NN,   DD, DD, 1, nullptr, nullptr, nullptr, Kb);
  gemm(LNSN, w_n2t_v, n2t_vb, NN,   DD, DD, 1, nullptr, nullptr, nullptr, Vb);
  attn(Qb, Kb, Vb, nt, nt + EC, EC, NTRI, SC);
  f32_to_bf16_k<<<1024, 256, 0, stream>>>(AGG, Qb, NTE);
  gemm(Qb, w_n2t_o, n2t_ob, NTRI, DD, DD, 0, out_tri, XLNf, out_tri, nullptr);

  // ---- tri MLP ----
  ln(out_tri, ln_tm_g, ln_tm_b, XLNb, nullptr, NTRI);
  gemm(XLNb, w_mt_1, mt_b1, NTRI, DD,  DFF, 2, nullptr, nullptr, nullptr, Hb);
  gemm(Hb,   w_mt_2, mt_b2, NTRI, DFF, DD,  0, out_tri, nullptr, out_tri, nullptr);
}
